// MyMultiheadAttention_10359461117985
// MI455X (gfx1250) — matrix operand order repaired and checked by reading, not compiled, not run on hardware
//
#include <hip/hip_runtime.h>

typedef __attribute__((ext_vector_type(16))) __bf16 v16bf;
typedef __attribute__((ext_vector_type(8)))  __bf16 v8bf;
typedef __attribute__((ext_vector_type(8)))  float  v8f;
typedef __attribute__((ext_vector_type(4)))  unsigned int v4u;
typedef __attribute__((ext_vector_type(8)))  int v8i;
typedef __attribute__((ext_vector_type(4)))  int v4i;

#define EDIM 4096
#define HEADS 32
#define HDIM 128
#define SEQ 17
#define BSZ 256
#define MTOK (SEQ*BSZ)       /* 4352 */
#define SC   (SEQ*SEQ)       /* 289  */
#define BH   (BSZ*HEADS)     /* 8192 */

// Tensor Data Mover availability (device pass only; host pass stays 0)
#if defined(__HIP_DEVICE_COMPILE__) && defined(__has_builtin)
#  if __has_builtin(__builtin_amdgcn_tensor_load_to_lds) && __has_builtin(__builtin_amdgcn_s_wait_tensorcnt)
#    define HAVE_TDM 1
#  endif
#endif
#ifndef HAVE_TDM
#  define HAVE_TDM 0
#endif

__device__ __forceinline__ unsigned short f2bf(float f) {
  union { float f; unsigned int u; } x; x.f = f;
  unsigned int r = x.u + 0x7FFFu + ((x.u >> 16) & 1u);   // round-to-nearest-even
  return (unsigned short)(r >> 16);
}
__device__ __forceinline__ float bf2f(unsigned short h) {
  union { unsigned int u; float f; } x; x.u = ((unsigned int)h) << 16;
  return x.f;
}

// ---------------------------------------------------------------- fp32 -> bf16
__global__ __launch_bounds__(256)
void k_f32_to_bf16(const float* __restrict__ src, unsigned short* __restrict__ dst, long n) {
  long i = ((long)blockIdx.x * 256 + threadIdx.x) * 4;
  long stride = (long)gridDim.x * 256 * 4;
  for (; i < n; i += stride) {
    float4 f = *(const float4*)(src + i);
    union { unsigned short s[4]; uint2 v; } o;
    o.s[0] = f2bf(f.x); o.s[1] = f2bf(f.y); o.s[2] = f2bf(f.z); o.s[3] = f2bf(f.w);
    *(uint2*)(dst + i) = o.v;
  }
}

// ---------------------------------------------------------------- WMMA GEMM
// C[M,N] = alpha * A[M,K](bf16) x W[N,K](bf16)^T + bias[N]
// Block tile 128x128, 8 waves; wave tile 32x64 = 2x4 wmma accumulators.
// TDM path: wave 0 DMAs whole 128x32 tiles into LDS (D# padding gives the
// 96B row stride), double-buffered, s_wait_tensorcnt + barrier sync.
#define TK 32
#define LDSW 48                         /* padded row stride in bf16 elems (96B) */
#define TILE_ELEMS (128 * LDSW)         /* 6144 elems = 12288 B per tile */
#define GEMM_LDS_BYTES (4 * TILE_ELEMS * 2)  /* 2 bufs x (A+B) = 49152 B */

#if HAVE_TDM
// Issue one 128-row x 32-elem bf16 tile load: global (row stride K elems) -> LDS
// with 8-DWORD padding after every 16 DWORDs (64B row -> 96B LDS stride).
__device__ __forceinline__ void tdm_issue(unsigned ldsoff_bytes,
                                          const unsigned short* gptr, int K) {
  unsigned long long ga = (unsigned long long)(size_t)gptr;
  v4u g0;
  g0[0] = 1u;                                    // count=1, user descriptor
  g0[1] = ldsoff_bytes;                          // lds_addr
  g0[2] = (unsigned)ga;                          // global_addr[31:0]
  g0[3] = (unsigned)((ga >> 32) & 0x1FFFFFFull) | (2u << 30);  // addr[56:32] | type=2
  v8i g1;
  g1[0] = (int)((1u << 16) |                     // data_size = 2B
                (1u << 20) |                     // pad_enable
                (3u << 22) |                     // pad_interval = 16 DWORDs
                (7u << 25));                     // pad_amount   = 8 DWORDs
  g1[1] = (int)(((unsigned)K & 0xFFFFu) << 16);  // tensor_dim0 lo16
  g1[2] = (int)(((unsigned)K >> 16) & 0xFFFFu);  // tensor_dim0 hi16 | dim1 lo16(=0)
  g1[3] = (int)(16u | (32u << 16));              // tensor_dim1 hi16 (1M rows) | tile_dim0=32
  g1[4] = 128;                                   // tile_dim1=128, tile_dim2=0
  g1[5] = K;                                     // tensor_dim0_stride lo32
  g1[6] = 0;
  g1[7] = 0;
  v4i g2; g2[0] = 1; g2[1] = 1; g2[2] = 0; g2[3] = 0;      // dim2=1, dim3=1
  v4i g3; g3[0] = 0; g3[1] = (1 << 16); g3[2] = 0; g3[3] = 0; // dim4=1
#if __clang_major__ >= 23
  v8i g4; g4[0]=0; g4[1]=0; g4[2]=0; g4[3]=0; g4[4]=0; g4[5]=0; g4[6]=0; g4[7]=0;
  __builtin_amdgcn_tensor_load_to_lds(g0, g1, g2, g3, g4, 0);
#else
  __builtin_amdgcn_tensor_load_to_lds(g0, g1, g2, g3, 0);
#endif
}
#endif

template <int OUT_BF16>
__global__ __launch_bounds__(256)
void k_gemm_bf16(const unsigned short* __restrict__ A,
                 const unsigned short* __restrict__ W,
                 void* __restrict__ Cout,
                 int M, int N, int K,
                 float alpha, const float* __restrict__ bias)
{
  extern __shared__ __align__(16) unsigned short smem[];

  const int tid  = threadIdx.x;
  const int wave = tid >> 5;
  const int lane = tid & 31;
  const int half = lane >> 4;
  const int lr   = lane & 15;
  const int mBase = blockIdx.y * 128;
  const int nBase = blockIdx.x * 128;
  const int wm = wave & 3;   // m strip of 32
  const int wn = wave >> 2;  // n strip of 64

  v8f acc[2][4];
#pragma unroll
  for (int i = 0; i < 2; ++i)
#pragma unroll
    for (int j = 0; j < 4; ++j) acc[i][j] = (v8f){0.f,0.f,0.f,0.f,0.f,0.f,0.f,0.f};

  const int nk = K / TK;

#if HAVE_TDM
  const unsigned tileBytes = TILE_ELEMS * 2;
  const unsigned short* Abase = A + (size_t)mBase * K;
  const unsigned short* Bbase = W + (size_t)nBase * K;
  if (wave == 0) {
    tdm_issue(0, Abase, K);
    tdm_issue(tileBytes, Bbase, K);
    if (nk > 1) {
      tdm_issue(2 * tileBytes, Abase + TK, K);
      tdm_issue(3 * tileBytes, Bbase + TK, K);
    }
  }
#endif

  for (int kt = 0; kt < nk; ++kt) {
#if HAVE_TDM
    const int buf = kt & 1;
    if (wave == 0) {
      if (kt + 1 < nk) __builtin_amdgcn_s_wait_tensorcnt(2);  // oldest pair done
      else             __builtin_amdgcn_s_wait_tensorcnt(0);
    }
    __syncthreads();
    const unsigned short* As_ = smem + (size_t)buf * 2 * TILE_ELEMS;
    const unsigned short* Bs_ = As_ + TILE_ELEMS;
#else
    const int k0 = kt * TK;
    unsigned short* As_ = smem;
    unsigned short* Bs_ = smem + TILE_ELEMS;
    __syncthreads();
#pragma unroll
    for (int i = 0; i < 2; ++i) {          // cooperative staging fallback
      int c   = tid + (i << 8);
      int row = c >> 2;
      int cc  = (c & 3) << 3;
      *(uint4*)(&As_[row * LDSW + cc]) =
          *(const uint4*)(A + (size_t)(mBase + row) * K + k0 + cc);
      *(uint4*)(&Bs_[row * LDSW + cc]) =
          *(const uint4*)(W + (size_t)(nBase + row) * K + k0 + cc);
    }
    __syncthreads();
#endif

    // A fragments: lane half h holds K = h*8..h*8+7 and 16+h*8..+7 (ISA 7.12.2)
    v16bf afrag[2];
#pragma unroll
    for (int mt = 0; mt < 2; ++mt) {
      const unsigned short* p = &As_[(wm * 32 + mt * 16 + lr) * LDSW + half * 8];
      v8bf lo = *(const v8bf*)p;
      v8bf hi = *(const v8bf*)(p + 16);
      afrag[mt] = __builtin_shufflevector(lo, hi, 0,1,2,3,4,5,6,7,8,9,10,11,12,13,14,15);
    }
    // B fragments: lane half h holds K = h*8..h*8+7 and 16+h*8..+7 for column lr
    v16bf bfrag[4];
#pragma unroll
    for (int nt = 0; nt < 4; ++nt) {
      const unsigned short* p = &Bs_[(wn * 64 + nt * 16 + lr) * LDSW + half * 8];
      v8bf lo = *(const v8bf*)p;
      v8bf hi = *(const v8bf*)(p + 16);
      bfrag[nt] = __builtin_shufflevector(lo, hi, 0,1,2,3,4,5,6,7,8,9,10,11,12,13,14,15);
    }
#pragma unroll
    for (int mt = 0; mt < 2; ++mt)
#pragma unroll
      for (int nt = 0; nt < 4; ++nt)
        acc[mt][nt] = __builtin_amdgcn_wmma_f32_16x16x32_bf16(
            false, afrag[mt], false, bfrag[nt], (short)0, acc[mt][nt], false, false);

#if HAVE_TDM
    __syncthreads();                        // all waves done reading buf
    if (wave == 0 && kt + 2 < nk) {
      tdm_issue((unsigned)(buf * 2) * tileBytes,             Abase + (kt + 2) * TK, K);
      tdm_issue((unsigned)(buf * 2) * tileBytes + tileBytes, Bbase + (kt + 2) * TK, K);
    }
#endif
  }

  // Epilogue (branch-free): lane: n = lr, VGPR i: m = half*8 + i
#pragma unroll
  for (int mt = 0; mt < 2; ++mt)
#pragma unroll
    for (int nt = 0; nt < 4; ++nt) {
      int col = nBase + wn * 64 + nt * 16 + lr;
      float bv = bias[col];
#pragma unroll
      for (int i = 0; i < 8; ++i) {
        int row = mBase + wm * 32 + mt * 16 + half * 8 + i;
        float v = acc[mt][nt][i] * alpha + bv;
        size_t idx = (size_t)row * N + col;
        if (OUT_BF16) ((unsigned short*)Cout)[idx] = f2bf(v);
        else          ((float*)Cout)[idx] = v;
      }
    }
}

// ---------------------------------------------------------------- qr_m / kr_m
__global__ __launch_bounds__(256)
void k_qrkr(const float* __restrict__ q0, const float* __restrict__ qr,
            const float* __restrict__ kr, float* __restrict__ qr_m,
            float* __restrict__ kr_m)
{
  const int d = blockIdx.x;          // 0..127
  const int tid = threadIdx.x;
  float aq[SEQ], ak[SEQ];
#pragma unroll
  for (int r = 0; r < SEQ; ++r) { aq[r] = 0.f; ak[r] = 0.f; }

  for (int bh = tid; bh < BH; bh += 256) {
    int b = bh >> 5, h = bh & 31;
    float u = q0[(size_t)b * EDIM + h * HDIM + d];
    const float* qrow = qr + (size_t)bh * SEQ;
    const float* krow = kr + (size_t)bh * SEQ;
#pragma unroll
    for (int r = 0; r < SEQ; ++r) { aq[r] += u * qrow[r]; ak[r] += u * krow[r]; }
  }
  __shared__ float red[256];
  for (int r = 0; r < SEQ; ++r) {
    red[tid] = aq[r]; __syncthreads();
    for (int s = 128; s > 0; s >>= 1) { if (tid < s) red[tid] += red[tid + s]; __syncthreads(); }
    if (tid == 0) qr_m[d * SEQ + r] = red[0];
    __syncthreads();
    red[tid] = ak[r]; __syncthreads();
    for (int s = 128; s > 0; s >>= 1) { if (tid < s) red[tid] += red[tid + s]; __syncthreads(); }
    if (tid == 0) kr_m[d * SEQ + r] = red[0];
    __syncthreads();
  }
}

// ---------------------------------------------------------------- zero helper
__global__ void k_zero_f32(float* p, int n) {
  int i = blockIdx.x * blockDim.x + threadIdx.x;
  if (i < n) p[i] = 0.f;
}

// ---------------------------------------------------------------- scores + BN sums
__global__ __launch_bounds__(256)
void k_scores(const unsigned short* __restrict__ Qb, const unsigned short* __restrict__ Kb,
              const float* __restrict__ qr_m, const float* __restrict__ kr_m,
              float* __restrict__ scores, float* __restrict__ bnsum, float* __restrict__ bnsum2)
{
  const int bh = blockIdx.x;
  const int b = bh >> 5, h = bh & 31;
  const int tid = threadIdx.x;
  __shared__ float qs[SEQ * HDIM], ks[SEQ * HDIM];
  __shared__ float qm[HDIM * SEQ], km[HDIM * SEQ];
  __shared__ float lsum[SEQ], lsum2[SEQ];

  for (int idx = tid; idx < SEQ * HDIM; idx += 256) {
    int l = idx >> 7, d = idx & 127;
    size_t off = ((size_t)(l * BSZ + b)) * EDIM + h * HDIM + d;
    qs[idx] = bf2f(Qb[off]);
    ks[idx] = bf2f(Kb[off]);
  }
  for (int idx = tid; idx < HDIM * SEQ; idx += 256) { qm[idx] = qr_m[idx]; km[idx] = kr_m[idx]; }
  if (tid < SEQ) { lsum[tid] = 0.f; lsum2[tid] = 0.f; }
  __syncthreads();

  for (int idx = tid; idx < SC; idx += 256) {
    int l = idx / SEQ, m = idx % SEQ;
    float s = 0.f;
    for (int d = 0; d < HDIM; ++d) {
      float qv = qs[l * HDIM + d];
      s += qv * ks[m * HDIM + d];               // q . k
      s += qv * qm[d * SEQ + m];                // q @ qr_m
      s += ks[l * HDIM + d] * km[d * SEQ + m];  // k @ kr_m (row l!)
    }
    scores[(size_t)bh * SC + idx] = s;
    atomicAdd(&lsum[l], s);
    atomicAdd(&lsum2[l], s * s);
  }
  __syncthreads();
  if (tid < SEQ) { atomicAdd(&bnsum[tid], lsum[tid]); atomicAdd(&bnsum2[tid], lsum2[tid]); }
}

// ---------------------------------------------------------------- BN finalize
__global__ void k_bnstats(const float* __restrict__ bnsum, const float* __restrict__ bnsum2,
                          float* __restrict__ bnmean, float* __restrict__ bnrstd)
{
  int l = threadIdx.x;
  if (l < SEQ) {
    const float cnt = (float)BH * (float)SEQ;
    float mean = bnsum[l] / cnt;
    float var  = bnsum2[l] / cnt - mean * mean;
    bnmean[l] = mean;
    bnrstd[l] = rsqrtf(var + 1e-5f);
  }
}

// ---------------------------------------------------------------- normalize+softmax+out+avg
__global__ __launch_bounds__(256)
void k_attn_out(const float* __restrict__ scores,
                const float* __restrict__ bnmean, const float* __restrict__ bnrstd,
                const unsigned short* __restrict__ Vb, const float* __restrict__ vr,
                unsigned short* __restrict__ AttnA, float* __restrict__ attn_avg)
{
  const int b = blockIdx.x;
  const int tid = threadIdx.x;
  __shared__ float w[SC];
  __shared__ float avg[SC];
  __shared__ float vsh[SEQ * HDIM];

  for (int idx = tid; idx < SC; idx += 256) avg[idx] = 0.f;
  __syncthreads();

  for (int h = 0; h < HEADS; ++h) {
    const int bh = b * HEADS + h;
    for (int idx = tid; idx < SC; idx += 256) {
      int l = idx / SEQ;
      w[idx] = (scores[(size_t)bh * SC + idx] - bnmean[l]) * bnrstd[l];
    }
    for (int idx = tid; idx < SEQ * HDIM; idx += 256) {
      int m = idx >> 7, d = idx & 127;
      vsh[idx] = bf2f(Vb[((size_t)(m * BSZ + b)) * EDIM + h * HDIM + d]) + vr[idx];
    }
    __syncthreads();
    if (tid < SEQ) {                       // per-row softmax
      int l = tid;
      float mx = -3.0e38f;
      for (int m = 0; m < SEQ; ++m) mx = fmaxf(mx, w[l * SEQ + m]);
      float s = 0.f;
      for (int m = 0; m < SEQ; ++m) { float e = __expf(w[l * SEQ + m] - mx); w[l * SEQ + m] = e; s += e; }
      float inv = 1.f / s;
      for (int m = 0; m < SEQ; ++m) w[l * SEQ + m] *= inv;
    }
    __syncthreads();
    for (int idx = tid; idx < SC; idx += 256) avg[idx] += w[idx] * (1.f / HEADS);
    for (int idx = tid; idx < SEQ * HDIM; idx += 256) {
      int l = idx >> 7, d = idx & 127;
      float o = 0.f;
#pragma unroll
      for (int m = 0; m < SEQ; ++m) o += w[l * SEQ + m] * vsh[m * HDIM + d];
      AttnA[((size_t)(l * BSZ + b)) * EDIM + h * HDIM + d] = f2bf(o);
    }
    __syncthreads();
  }
  for (int idx = tid; idx < SC; idx += 256) attn_avg[(size_t)b * SC + idx] = avg[idx];
}

// ---------------------------------------------------------------- launch
extern "C" void kernel_launch(void* const* d_in, const int* in_sizes, int n_in,
                              void* d_out, int out_size, void* d_ws, size_t ws_size,
                              hipStream_t stream) {
  const float* query = (const float*)d_in[0];
  const float* key_  = (const float*)d_in[1];
  const float* value = (const float*)d_in[2];
  const float* Wqf   = (const float*)d_in[3];
  const float* Wkf   = (const float*)d_in[4];
  const float* Wvf   = (const float*)d_in[5];
  const float* Wof   = (const float*)d_in[6];
  const float* bias  = (const float*)d_in[7];
  const float* qr    = (const float*)d_in[8];
  const float* kr    = (const float*)d_in[9];
  const float* vr    = (const float*)d_in[10];
  float* out = (float*)d_out;

  char* ws = (char*)d_ws;
  size_t off = 0;
  auto take = [&](size_t bytes) -> char* {
    char* p = ws + off; off += (bytes + 255) & ~(size_t)255; return p;
  };
  const size_t Wsz = (size_t)EDIM * EDIM;
  const size_t Xsz = (size_t)MTOK * EDIM;
  unsigned short* Wq = (unsigned short*)take(Wsz * 2);
  unsigned short* Wk = (unsigned short*)take(Wsz * 2);
  unsigned short* Wv = (unsigned short*)take(Wsz * 2);
  unsigned short* Wo = (unsigned short*)take(Wsz * 2);
  unsigned short* Xq = (unsigned short*)take(Xsz * 2);
  unsigned short* Xk = (unsigned short*)take(Xsz * 2);
  unsigned short* Xv = (unsigned short*)take(Xsz * 2);
  unsigned short* Qb = (unsigned short*)take(Xsz * 2);
  unsigned short* Kb = (unsigned short*)take(Xsz * 2);
  unsigned short* Vb = (unsigned short*)take(Xsz * 2);
  unsigned short* Aout = (unsigned short*)take(Xsz * 2);
  float* scores = (float*)take((size_t)BH * SC * 4);
  float* qr_m   = (float*)take((size_t)HDIM * SEQ * 4);
  float* kr_m   = (float*)take((size_t)HDIM * SEQ * 4);
  float* zbias  = (float*)take((size_t)EDIM * 4);
  float* bnsum  = (float*)take(4 * SEQ * 4);
  float* bnsum2 = bnsum + SEQ;
  float* bnmean = bnsum + 2 * SEQ;
  float* bnrstd = bnsum + 3 * SEQ;

  // 1) bf16 conversions + zero fills
  k_f32_to_bf16<<<2048, 256, 0, stream>>>(Wqf, Wq, (long)Wsz);
  k_f32_to_bf16<<<2048, 256, 0, stream>>>(Wkf, Wk, (long)Wsz);
  k_f32_to_bf16<<<2048, 256, 0, stream>>>(Wvf, Wv, (long)Wsz);
  k_f32_to_bf16<<<2048, 256, 0, stream>>>(Wof, Wo, (long)Wsz);
  k_f32_to_bf16<<<2048, 256, 0, stream>>>(query, Xq, (long)Xsz);
  k_f32_to_bf16<<<2048, 256, 0, stream>>>(key_,  Xk, (long)Xsz);
  k_f32_to_bf16<<<2048, 256, 0, stream>>>(value, Xv, (long)Xsz);
  k_zero_f32<<<16, 256, 0, stream>>>(zbias, EDIM);
  k_zero_f32<<<1, 128, 0, stream>>>(bnsum, 4 * SEQ);

  // 2) Q/K/V projections (WMMA); q pre-scaled by HD^-0.5
  dim3 gg(EDIM / 128, MTOK / 128);   // (32, 34)
  const float scaling = 0.08838834764831845f;
  k_gemm_bf16<1><<<gg, 256, GEMM_LDS_BYTES, stream>>>(Xq, Wq, Qb, MTOK, EDIM, EDIM, scaling, zbias);
  k_gemm_bf16<1><<<gg, 256, GEMM_LDS_BYTES, stream>>>(Xk, Wk, Kb, MTOK, EDIM, EDIM, 1.0f, zbias);
  k_gemm_bf16<1><<<gg, 256, GEMM_LDS_BYTES, stream>>>(Xv, Wv, Vb, MTOK, EDIM, EDIM, 1.0f, zbias);

  // 3) relative-position matrices from raw query[0]
  k_qrkr<<<HDIM, 256, 0, stream>>>(query, qr, kr, qr_m, kr_m);

  // 4) scores + BatchNorm
  k_scores<<<BH, 256, 0, stream>>>(Qb, Kb, qr_m, kr_m, scores, bnsum, bnsum2);
  k_bnstats<<<1, 32, 0, stream>>>(bnsum, bnsum2, bnmean, bnrstd);

  // 5) normalize + softmax + w@(v+vr) + attn_avg
  float* attn_avg = out + (size_t)MTOK * EDIM;
  k_attn_out<<<BSZ, 256, 0, stream>>>(scores, bnmean, bnrstd, Vb, vr, Aout, attn_avg);

  // 6) output projection with bias -> Z (fp32, start of d_out)
  k_gemm_bf16<0><<<gg, 256, GEMM_LDS_BYTES, stream>>>(Aout, Wo, out, MTOK, EDIM, EDIM, 1.0f, bias);
}
